// ASSM_35648228557255
// MI455X (gfx1250) — matrix operand order repaired and checked by reading, not compiled, not run on hardware
//
#include <hip/hip_runtime.h>
#include <math.h>

// ---------------- problem constants (match reference setup) ----------------
#define BSZ   16
#define LSEQ  4096
#define TTOK  (BSZ * LSEQ)   // 65536 tokens
#define DIM_  192
#define HID_  384
#define DST_  16
#define NT_   17
#define IR_   128
#define DTR_  24

typedef __attribute__((ext_vector_type(16))) __bf16 v16bf;
typedef __attribute__((ext_vector_type(8)))  float  v8f;

// ---------------- WMMA fragment helpers (bf16 A/B, f32 C/D) ----------------
// A-matrix 16x32 (16-bit): lanes 0-15 -> M=lane, K-chunks {0..7, 16..23};
//                          lanes 16-31 -> M=lane-16, K-chunks {8..15, 24..31}.
__device__ __forceinline__ v16bf load_a_f32(const float* src, int ld, int k0, int kvalid) {
  int lane = threadIdx.x & 31;
  int m = lane & 15;
  int kb = (lane >> 4) << 3;
  const float* p = src + (size_t)m * ld + k0;
  v16bf r;
#pragma unroll
  for (int j = 0; j < 8; ++j) {
    int k1 = kb + j, k2 = kb + 16 + j;
    r[j]     = (k1 < kvalid) ? (__bf16)p[k1] : (__bf16)0.0f;
    r[j + 8] = (k2 < kvalid) ? (__bf16)p[k2] : (__bf16)0.0f;
  }
  return r;
}

__device__ __forceinline__ v16bf load_a_bf16(const __bf16* src, int ld, int k0) {
  int lane = threadIdx.x & 31;
  int m = lane & 15;
  int kb = (lane >> 4) << 3;
  const __bf16* p = src + (size_t)m * ld + k0;
  v16bf r;
#pragma unroll
  for (int j = 0; j < 8; ++j) { r[j] = p[kb + j]; r[j + 8] = p[kb + 16 + j]; }
  return r;
}

// B-matrix 32x16 (16-bit): lanes 0-15 hold K={0..7, 16..23} of column N=lane;
//                          lanes 16-31 hold K={8..15, 24..31} of column N=lane-16.
// W is row-major (Nout, K); B(k, n) = W[n][k].
__device__ __forceinline__ v16bf load_b_f32(const float* W, int ldk, int n0, int nmax,
                                            int k0, int kvalid) {
  int lane = threadIdx.x & 31;
  int n = n0 + (lane & 15);
  int ks = (lane >> 4) << 3;  // 0 or 8
  bool nok = (n < nmax);
  const float* p = W + (size_t)n * ldk + k0;
  v16bf r;
#pragma unroll
  for (int j = 0; j < 16; ++j) {
    int kk = ks + (j & 7) + ((j >> 3) << 4);
    r[j] = (nok && kk < kvalid) ? (__bf16)p[kk] : (__bf16)0.0f;
  }
  return r;
}

__device__ __forceinline__ v8f wmma_bf16(v16bf a, v16bf b, v8f c) {
  return __builtin_amdgcn_wmma_f32_16x16x32_bf16(false, a, false, b, (short)0, c, false, false);
}

// C/D 16x16 f32: lanes 0-15 -> N=lane, M=r; lanes 16-31 -> N=lane-16, M=8+r.
__device__ __forceinline__ void cpos(int& mb, int& n) {
  int lane = threadIdx.x & 31;
  n = lane & 15;
  mb = (lane >> 4) << 3;
}

// ---------------- K0: full_emb = embB @ token_weight (17x16) ----------------
__global__ void k_full_emb(const float* __restrict__ embB, const float* __restrict__ tw,
                           float* __restrict__ fe) {
  int t = threadIdx.x;
  if (t < NT_ * DST_) {
    int i = t / DST_, j = t % DST_;
    float s = 0.f;
    for (int k = 0; k < IR_; ++k) s += embB[i * IR_ + k] * tw[k * DST_ + j];
    fe[t] = s;
  }
}

// ---------------- K1: routing MLP + gumbel argmax -> idx --------------------
// argmax(softmax(log_softmax(z)+g)) == argmax(z+g)  (per-token constants drop)
__global__ __launch_bounds__(32) void k_route(const float* __restrict__ x,
                                              const float* __restrict__ w1,
                                              const float* __restrict__ b1,
                                              const float* __restrict__ w2,
                                              const float* __restrict__ b2,
                                              const float* __restrict__ gum,
                                              int* __restrict__ idxOut) {
  __shared__ float hm[16][64];
  int tile = blockIdx.x;
  const float* xA = x + (size_t)tile * 16 * DIM_;
  v8f c[4] = {};
  for (int kk = 0; kk < DIM_; kk += 32) {
    v16bf a = load_a_f32(xA, DIM_, kk, 32);
#pragma unroll
    for (int nc = 0; nc < 4; ++nc) {
      v16bf b = load_b_f32(w1, DIM_, nc * 16, 64, kk, 32);
      c[nc] = wmma_bf16(a, b, c[nc]);
    }
  }
  int mb, n; cpos(mb, n);
#pragma unroll
  for (int nc = 0; nc < 4; ++nc) {
    int col = nc * 16 + n;
    float bias = b1[col];
#pragma unroll
    for (int r = 0; r < 8; ++r) {
      float v = c[nc][r] + bias;
      float v3 = v * v * v;  // gelu (tanh approximation, jax default)
      hm[mb + r][col] = 0.5f * v * (1.f + tanhf(0.7978845608028654f * (v + 0.044715f * v3)));
    }
  }
  __syncthreads();
  int lane = threadIdx.x;
  int m = lane & 15;
  int token = tile * 16 + m;
  const float* gp = gum + (size_t)token * NT_;
  int c0 = (lane < 16) ? 0 : 9;
  int cn = (lane < 16) ? 9 : 8;
  float best = -1e30f; int bi = 0x7fffffff;
  for (int cc = 0; cc < cn; ++cc) {
    int cls = c0 + cc;
    float s = b2[cls];
#pragma unroll 8
    for (int k = 0; k < 64; ++k) s += hm[m][k] * w2[cls * 64 + k];
    s += gp[cls];
    if (s > best) { best = s; bi = cls; }
  }
  float ob = __shfl_xor(best, 16, 32);
  int   oi = __shfl_xor(bi, 16, 32);
  if (ob > best || (ob == best && oi < bi)) { best = ob; bi = oi; }
  if (lane < 16) idxOut[token] = bi;
}

// ---------------- K2: stable counting sort per batch ------------------------
__global__ __launch_bounds__(256) void k_sort(const int* __restrict__ idx,
                                              int* __restrict__ sortIdx) {
  __shared__ int cnt[256][NT_];
  __shared__ int offs[256][NT_];
  int b = blockIdx.x, t = threadIdx.x;
  const int* ip = idx + (size_t)b * LSEQ;
  int base = t * 16;
  for (int c = 0; c < NT_; ++c) cnt[t][c] = 0;
  for (int i = 0; i < 16; ++i) cnt[t][ip[base + i]]++;
  __syncthreads();
  if (t == 0) {
    int run = 0;
    for (int c = 0; c < NT_; ++c)
      for (int th = 0; th < 256; ++th) { offs[th][c] = run; run += cnt[th][c]; }
  }
  __syncthreads();
  int* so = sortIdx + (size_t)b * LSEQ;
  for (int i = 0; i < 16; ++i) {
    int tok = base + i;
    int c = ip[tok];
    int p = offs[t][c]++;
    so[p] = tok;
  }
}

// ---------------- K3: in_proj GEMM (T,192)x(192,384) + bias -----------------
__global__ __launch_bounds__(128) void k_inproj(const float* __restrict__ x,
                                                const float* __restrict__ w,
                                                const float* __restrict__ bias,
                                                float* __restrict__ out) {
  int wave = threadIdx.x >> 5;
  int mt = blockIdx.x;
  int nt = blockIdx.y * 4 + wave;  // 0..23
  const float* xA = x + (size_t)mt * 16 * DIM_;
  v8f c = {};
  for (int kk = 0; kk < DIM_; kk += 32) {
    v16bf a = load_a_f32(xA, DIM_, kk, 32);
    v16bf b = load_b_f32(w, DIM_, nt * 16, HID_, kk, 32);
    c = wmma_bf16(a, b, c);
  }
  int mb, n; cpos(mb, n);
  int col = nt * 16 + n;
  float bv = bias[col];
#pragma unroll
  for (int r = 0; r < 8; ++r)
    out[((size_t)(mt * 16 + mb + r)) * HID_ + col] = c[r] + bv;
}

// ---------------- K4: depthwise 3x3 CPE + sigmoid gate ----------------------
__global__ __launch_bounds__(256) void k_cpe(const float* __restrict__ xi,
                                             const float* __restrict__ cw,
                                             const float* __restrict__ cb,
                                             const int* __restrict__ Hp,
                                             const int* __restrict__ Wp,
                                             float* __restrict__ out) {
  size_t id = (size_t)blockIdx.x * 256 + threadIdx.x;
  if (id >= (size_t)TTOK * HID_) return;
  int c = (int)(id % HID_);
  size_t tl = id / HID_;
  int l = (int)(tl % LSEQ);
  int b = (int)(tl / LSEQ);
  int H = *Hp, W = *Wp;
  int h = l / W, w = l % W;
  float acc = cb[c];
  const float* wp = cw + c * 9;
#pragma unroll
  for (int dh = -1; dh <= 1; ++dh) {
    int hh = h + dh;
    if (hh < 0 || hh >= H) continue;
#pragma unroll
    for (int dw = -1; dw <= 1; ++dw) {
      int ww = w + dw;
      if (ww < 0 || ww >= W) continue;
      acc += wp[(dh + 1) * 3 + (dw + 1)] * xi[((size_t)b * LSEQ + hh * W + ww) * HID_ + c];
    }
  }
  float v = xi[id];
  out[id] = v * (1.f / (1.f + __expf(-acc)));
}

// ---------------- K5: gather rows by sort_idx -------------------------------
__global__ __launch_bounds__(256) void k_gather(const float* __restrict__ gated,
                                                const int* __restrict__ sortIdx,
                                                float* __restrict__ sem) {
  size_t id = (size_t)blockIdx.x * 256 + threadIdx.x;
  if (id >= (size_t)TTOK * HID_) return;
  int c = (int)(id % HID_);
  size_t tl = id / HID_;
  int b = (int)(tl / LSEQ);
  int src = sortIdx[tl];
  sem[id] = gated[((size_t)b * LSEQ + src) * HID_ + c];
}

// ---------------- K6: x_proj GEMM (T,384)x(384,56) -> dts_r / Bs / Cs -------
__global__ __launch_bounds__(128) void k_xproj(const float* __restrict__ sem,
                                               const float* __restrict__ w,
                                               float* __restrict__ dtsR,
                                               float* __restrict__ Bs,
                                               float* __restrict__ Cs) {
  int nt = threadIdx.x >> 5;   // 4 N-tiles cover 64 (56 valid)
  int mt = blockIdx.x;
  const float* A = sem + (size_t)mt * 16 * HID_;
  v8f c = {};
  for (int kk = 0; kk < HID_; kk += 32) {
    v16bf a = load_a_f32(A, HID_, kk, 32);
    v16bf b = load_b_f32(w, HID_, nt * 16, DTR_ + 2 * DST_, kk, 32);
    c = wmma_bf16(a, b, c);
  }
  int mb, n; cpos(mb, n);
  int col = nt * 16 + n;
#pragma unroll
  for (int r = 0; r < 8; ++r) {
    size_t tok = (size_t)mt * 16 + mb + r;
    float v = c[r];
    if (col < DTR_)                dtsR[tok * DTR_ + col] = v;
    else if (col < DTR_ + DST_)    Bs[tok * DST_ + (col - DTR_)] = v;
    else if (col < DTR_ + 2*DST_)  Cs[tok * DST_ + (col - DTR_ - DST_)] = v;
  }
}

// ---------------- K7: dt GEMM (T,24)x(24,384) + softplus(.+bias) ------------
__global__ __launch_bounds__(128) void k_dt(const float* __restrict__ dtsR,
                                            const float* __restrict__ dtw,
                                            const float* __restrict__ dtb,
                                            float* __restrict__ delta) {
  int wave = threadIdx.x >> 5;
  int mt = blockIdx.x;
  int nt = blockIdx.y * 4 + wave;  // 0..23
  v8f c = {};
  v16bf a = load_a_f32(dtsR + (size_t)mt * 16 * DTR_, DTR_, 0, DTR_);  // K padded to 32
  v16bf b = load_b_f32(dtw, DTR_, nt * 16, HID_, 0, DTR_);
  c = wmma_bf16(a, b, c);
  int mb, n; cpos(mb, n);
  int col = nt * 16 + n;
  float bv = dtb[col];
#pragma unroll
  for (int r = 0; r < 8; ++r) {
    float v = c[r] + bv;
    v = (v > 20.f) ? v : log1pf(__expf(v));  // softplus
    delta[((size_t)mt * 16 + mb + r) * HID_ + col] = v;
  }
}

// ---------------- K8: selective scan (sequential over L) --------------------
// Block = 256 threads = 16 channel-rows x 16 state-lanes. grid = (HID/16, B).
__global__ __launch_bounds__(256) void k_scan(const float* __restrict__ u,
                                              const float* __restrict__ delta,
                                              const float* __restrict__ Bs,
                                              const float* __restrict__ Cs,
                                              const int* __restrict__ idx,
                                              const float* __restrict__ fullEmb,
                                              const float* __restrict__ A_logs,
                                              const float* __restrict__ Ds,
                                              const int* __restrict__ gtp,
                                              float* __restrict__ y) {
  int n = threadIdx.x & 15;
  int d = blockIdx.x * 16 + (threadIdx.x >> 4);
  int b = blockIdx.y;
  int gt = *gtp;
  float A = -__expf(A_logs[d * DST_ + n]);
  float Dv = Ds[d];
  float h = 0.f;
  size_t baseTok = (size_t)b * LSEQ;
  for (int l = 0; l < LSEQ; ++l) {
    size_t tok = baseTok + l;
    if (l + 8 < LSEQ) {  // keep the row streams warm
      __builtin_prefetch(&delta[(tok + 8) * HID_ + d], 0, 1);
      __builtin_prefetch(&u[(tok + 8) * HID_ + d], 0, 1);
    }
    float dl = delta[tok * HID_ + d];
    float uv = u[tok * HID_ + d];
    float bn = Bs[tok * DST_ + n];
    float cn = Cs[tok * DST_ + n];
    if (gt != 0) cn += fullEmb[idx[tok] * DST_ + n];  // reference adds original-order prompt
    float da = __expf(dl * A);
    h = da * h + (dl * uv) * bn;
    float yp = h * cn;
    yp += __shfl_xor(yp, 1, 16);
    yp += __shfl_xor(yp, 2, 16);
    yp += __shfl_xor(yp, 4, 16);
    yp += __shfl_xor(yp, 8, 16);
    if (n == 0) y[tok * HID_ + d] = yp + uv * Dv;
  }
}

// ---------------- K9: LayerNorm over hid, store bf16 ------------------------
__global__ __launch_bounds__(128) void k_ln(const float* __restrict__ y,
                                            const float* __restrict__ g,
                                            const float* __restrict__ be,
                                            __bf16* __restrict__ yn) {
  __shared__ float ws1[4], ws2[4];
  size_t tok = blockIdx.x;
  int t = threadIdx.x;
  const float* yp = y + tok * HID_;
  float s = 0.f, s2 = 0.f;
  for (int c = t; c < HID_; c += 128) { float v = yp[c]; s += v; s2 += v * v; }
#pragma unroll
  for (int o = 1; o < 32; o <<= 1) { s += __shfl_xor(s, o, 32); s2 += __shfl_xor(s2, o, 32); }
  int wv = t >> 5;
  if ((t & 31) == 0) { ws1[wv] = s; ws2[wv] = s2; }
  __syncthreads();
  s  = ws1[0] + ws1[1] + ws1[2] + ws1[3];
  s2 = ws2[0] + ws2[1] + ws2[2] + ws2[3];
  float mu = s * (1.f / HID_);
  float var = s2 * (1.f / HID_) - mu * mu;
  float rstd = rsqrtf(var + 1e-5f);
  for (int c = t; c < HID_; c += 128)
    yn[tok * HID_ + c] = (__bf16)((yp[c] - mu) * rstd * g[c] + be[c]);
}

// ---------------- K10: out_proj GEMM + bias + scatter by sort_idx -----------
__global__ __launch_bounds__(128) void k_outproj(const __bf16* __restrict__ yn,
                                                 const float* __restrict__ w,
                                                 const float* __restrict__ bias,
                                                 const int* __restrict__ sortIdx,
                                                 float* __restrict__ out) {
  int wave = threadIdx.x >> 5;
  int mt = blockIdx.x;
  int nt = blockIdx.y * 4 + wave;  // 0..11
  const __bf16* A = yn + (size_t)mt * 16 * HID_;
  v8f c = {};
  for (int kk = 0; kk < HID_; kk += 32) {
    v16bf a = load_a_bf16(A, HID_, kk);
    v16bf b = load_b_f32(w, HID_, nt * 16, DIM_, kk, 32);
    c = wmma_bf16(a, b, c);
  }
  int mb, n; cpos(mb, n);
  int col = nt * 16 + n;
  float bv = bias[col];
#pragma unroll
  for (int r = 0; r < 8; ++r) {
    size_t p = (size_t)mt * 16 + mb + r;      // sorted position (global)
    int b_ = (int)(p / LSEQ);
    int dst = sortIdx[p];                     // original token index within batch
    out[((size_t)b_ * LSEQ + dst) * DIM_ + col] = c[r] + bv;
  }
}

// ---------------- launcher --------------------------------------------------
extern "C" void kernel_launch(void* const* d_in, const int* in_sizes, int n_in,
                              void* d_out, int out_size, void* d_ws, size_t ws_size,
                              hipStream_t stream) {
  const float* x         = (const float*)d_in[0];
  const float* token_w   = (const float*)d_in[1];
  const float* gumbel    = (const float*)d_in[2];
  const float* embB      = (const float*)d_in[3];
  const float* route_w1  = (const float*)d_in[4];
  const float* route_b1  = (const float*)d_in[5];
  const float* route_w2  = (const float*)d_in[6];
  const float* route_b2  = (const float*)d_in[7];
  const float* inproj_w  = (const float*)d_in[8];
  const float* inproj_b  = (const float*)d_in[9];
  const float* cpe_w     = (const float*)d_in[10];
  const float* cpe_b     = (const float*)d_in[11];
  const float* x_proj_w  = (const float*)d_in[12];
  const float* dt_w      = (const float*)d_in[13];
  const float* dt_b      = (const float*)d_in[14];
  const float* A_logs    = (const float*)d_in[15];
  const float* Ds        = (const float*)d_in[16];
  const float* norm_g    = (const float*)d_in[17];
  const float* norm_b    = (const float*)d_in[18];
  const float* outproj_w = (const float*)d_in[19];
  const float* outproj_b = (const float*)d_in[20];
  const int*   Hp        = (const int*)d_in[21];
  const int*   Wp        = (const int*)d_in[22];
  const int*   gtp       = (const int*)d_in[23];
  float* out = (float*)d_out;

  char* ws = (char*)d_ws;
  size_t off = 0;
  auto alloc = [&](size_t bytes) {
    off = (off + 255) & ~(size_t)255;
    size_t r = off;
    off += bytes;
    return r;
  };
  int*   idx     = (int*)(ws + alloc((size_t)TTOK * 4));
  int*   sortIdx = (int*)(ws + alloc((size_t)TTOK * 4));
  float* fullEmb = (float*)(ws + alloc((size_t)NT_ * DST_ * 4));
  float* dtsR    = (float*)(ws + alloc((size_t)TTOK * DTR_ * 4));
  float* Bsb     = (float*)(ws + alloc((size_t)TTOK * DST_ * 4));
  float* Csb     = (float*)(ws + alloc((size_t)TTOK * DST_ * 4));
  float* buf1    = (float*)(ws + alloc((size_t)TTOK * HID_ * 4));  // xi, later y
  float* buf2    = (float*)(ws + alloc((size_t)TTOK * HID_ * 4));  // gated, later yn(bf16)
  float* buf3    = (float*)(ws + alloc((size_t)TTOK * HID_ * 4));  // sem (= u for scan)
  float* buf4    = (float*)(ws + alloc((size_t)TTOK * HID_ * 4));  // delta

  const int MT = TTOK / 16;                 // 4096 M-tiles
  const size_t NELT = (size_t)TTOK * HID_;
  const int EB = (int)((NELT + 255) / 256);

  k_full_emb<<<1, 288, 0, stream>>>(embB, token_w, fullEmb);
  k_route<<<MT, 32, 0, stream>>>(x, route_w1, route_b1, route_w2, route_b2, gumbel, idx);
  k_sort<<<BSZ, 256, 0, stream>>>(idx, sortIdx);
  k_inproj<<<dim3(MT, HID_ / 64), 128, 0, stream>>>(x, inproj_w, inproj_b, buf1);
  k_cpe<<<EB, 256, 0, stream>>>(buf1, cpe_w, cpe_b, Hp, Wp, buf2);
  k_gather<<<EB, 256, 0, stream>>>(buf2, sortIdx, buf3);
  k_xproj<<<MT, 128, 0, stream>>>(buf3, x_proj_w, dtsR, Bsb, Csb);
  k_dt<<<dim3(MT, HID_ / 64), 128, 0, stream>>>(dtsR, dt_w, dt_b, buf4);
  k_scan<<<dim3(HID_ / 16, BSZ), 256, 0, stream>>>(buf3, buf4, Bsb, Csb, idx, fullEmb,
                                                   A_logs, Ds, gtp, buf1);
  k_ln<<<TTOK, 128, 0, stream>>>(buf1, norm_g, norm_b, (__bf16*)buf2);
  k_outproj<<<dim3(MT, DIM_ / 64), 128, 0, stream>>>((__bf16*)buf2, outproj_w, outproj_b,
                                                     sortIdx, out);
  (void)in_sizes; (void)n_in; (void)out_size; (void)ws_size;
}
